// GPT_75952201662852
// MI455X (gfx1250) — matrix operand order repaired and checked by reading, not compiled, not run on hardware
//
#include <hip/hip_runtime.h>
#include <stdint.h>

// Model dims
#define VV  32000
#define DD  768
#define LL  6
#define HH  12
#define HSZ 64
#define BB  4
#define TT  1024
#define BT  (BB*TT)
#define D3  (3*DD)
#define D4  (4*DD)

typedef __bf16 bf16;
typedef __bf16 v16bf __attribute__((ext_vector_type(16)));
typedef __bf16 v8bf  __attribute__((ext_vector_type(8)));
typedef float  v8f   __attribute__((ext_vector_type(8)));
typedef unsigned int u32x4 __attribute__((ext_vector_type(4)));
typedef int i32x8 __attribute__((ext_vector_type(8)));
typedef int i32x4 __attribute__((ext_vector_type(4)));

#if defined(__has_builtin)
#if __has_builtin(__builtin_amdgcn_tensor_load_to_lds) && __has_builtin(__builtin_amdgcn_s_wait_tensorcnt)
#define HAVE_TDM 1
#endif
#endif

static __device__ inline v8f zero8() {
  v8f z = {0.f,0.f,0.f,0.f,0.f,0.f,0.f,0.f};
  return z;
}

// D = A(16x32 bf16) * B(32x16 bf16) + C(16x16 f32)
static __device__ inline v8f wmma_bf16(v16bf a, v16bf b, v8f c) {
  return __builtin_amdgcn_wmma_f32_16x16x32_bf16(false, a, false, b, (short)0, c, false, false);
}

// A-matrix fragment pattern used here (16-bit A 16x32, contiguous k order):
// lane m = lane&15, elem j -> k = hi*16+j (callers pass lo = row+hi*16, hi = lo+8)
static __device__ inline v16bf frag_split(const bf16* lo, const bf16* hi) {
  v8bf l = *(const v8bf*)lo;
  v8bf h = *(const v8bf*)hi;
  v16bf r;
  #pragma unroll
  for (int i = 0; i < 8; i++) { r[i] = l[i]; r[i+8] = h[i]; }
  return r;
}

#ifdef HAVE_TDM
// Issue one 2D TDM load: tile (tile_w x tile_h) of bf16 from row-major global
// (row stride = stride elems) into contiguous LDS at lds_off. Per CDNA5 D# layout.
// This toolchain's builtin takes 6 args (g0, g1, g2, g3, g4-pad, cpol).
static __device__ inline void tdm_load_2d(unsigned lds_off, const void* gptr,
                                          unsigned tile_w, unsigned tile_h,
                                          unsigned tensor_w, unsigned tensor_h,
                                          unsigned stride) {
  unsigned long long ga = (unsigned long long)(size_t)gptr;
  u32x4 g0;
  g0[0] = 1u;                                        // count=1 (valid), load, user
  g0[1] = lds_off;                                   // lds_addr [63:32]
  g0[2] = (unsigned)(ga & 0xffffffffu);              // global_addr lo
  g0[3] = (unsigned)((ga >> 32) & 0x01ffffffu)       // global_addr hi (57-bit)
          | 0x80000000u;                             // type = 2 ("image")
  i32x8 g1;
  g1[0] = (int)(1u << 16);                           // wg_mask=0, data_size=1 (2B)
  g1[1] = (int)((tensor_w & 0xffffu) << 16);         // abar=0 | tensor_dim0[15:0]
  g1[2] = (int)((tensor_w >> 16) | ((tensor_h & 0xffffu) << 16)); // dim0 hi | dim1 lo
  g1[3] = (int)((tensor_h >> 16) | (tile_w << 16));  // dim1 hi | tile_dim0
  g1[4] = (int)tile_h;                               // tile_dim1 | tile_dim2=0
  g1[5] = (int)stride;                               // tensor_dim0_stride lo
  g1[6] = 0;                                         // stride hi | dim1_stride lo
  g1[7] = 0;
  i32x4 z4 = {0, 0, 0, 0};
  i32x8 z8 = {0, 0, 0, 0, 0, 0, 0, 0};
  __builtin_amdgcn_tensor_load_to_lds(g0, g1, z4, z4, z8, 0);
}
#endif

// ---------------- weight conversion ----------------

__global__ void cvt_flat(const float* __restrict__ s, bf16* __restrict__ d, int n) {
  int i = blockIdx.x * 256 + threadIdx.x;
  if (i < n) d[i] = (bf16)s[i];
}

// Wq/Wk/Wv [L,H,D,HS] fp32 -> wqkv [L][D][3*D] bf16 (col = off + h*HS + k)
__global__ void cvt_qkv(const float* __restrict__ s, bf16* __restrict__ d, int off) {
  int i = blockIdx.x * 256 + threadIdx.x;
  int k  = i % HSZ; int t = i / HSZ;
  int dd = t % DD;  t /= DD;
  int hh = t % HH;  int l = t / HH;
  d[(size_t)l*DD*D3 + (size_t)dd*D3 + off + hh*HSZ + k] = (bf16)s[i];
}

// ---------------- embedding ----------------

__global__ void embed_kernel(const int* __restrict__ idx, const float* __restrict__ tok,
                             const float* __restrict__ pos, float* __restrict__ x) {
  int row = blockIdx.x;
  int t = row % TT;
  int token = idx[row];
  for (int i = threadIdx.x; i < DD; i += 256)
    x[(size_t)row*DD + i] = tok[(size_t)token*DD + i] + pos[(size_t)t*DD + i];
}

// ---------------- layernorm (fp32 in, bf16 out) ----------------

__global__ void __launch_bounds__(256) ln_kernel(const float* __restrict__ x,
                                                 const float* __restrict__ g,
                                                 const float* __restrict__ b,
                                                 bf16* __restrict__ out) {
  __shared__ float red[256];
  int row = blockIdx.x, tid = threadIdx.x;
  const float* xr = x + (size_t)row*DD;
  float v0 = xr[tid], v1 = xr[tid+256], v2 = xr[tid+512];
  red[tid] = v0 + v1 + v2;
  __syncthreads();
  for (int o = 128; o > 0; o >>= 1) { if (tid < o) red[tid] += red[tid+o]; __syncthreads(); }
  float mean = red[0] * (1.0f/DD);
  __syncthreads();
  float d0 = v0-mean, d1 = v1-mean, d2 = v2-mean;
  red[tid] = d0*d0 + d1*d1 + d2*d2;
  __syncthreads();
  for (int o = 128; o > 0; o >>= 1) { if (tid < o) red[tid] += red[tid+o]; __syncthreads(); }
  float rs = rsqrtf(red[0] * (1.0f/DD) + 1e-5f);
  bf16* orow = out + (size_t)row*DD;
  orow[tid]     = (bf16)(d0*rs*g[tid]     + b[tid]);
  orow[tid+256] = (bf16)(d1*rs*g[tid+256] + b[tid+256]);
  orow[tid+512] = (bf16)(d2*rs*g[tid+512] + b[tid+512]);
}

// ---------------- WMMA GEMM: C[M,N] = act(A[M,K]*B[K,N] + bias (+res)) ----------------
// block tile 128x128, BK=32; 8 waves, each wave 32x64 (2x4 WMMA tiles).
// A tile (128x32, row-major in LDS) is DMA'd by the Tensor Data Mover (wave 0);
// B tile (32x128) is cooperatively loaded + transposed by all waves in parallel.

template<bool RELU, bool RES, bool F32OUT, bool BF16OUT>
__global__ void __launch_bounds__(256) gemm_kernel(
    const bf16* __restrict__ A, const bf16* __restrict__ Bm,
    const float* __restrict__ bias, const float* __restrict__ res,
    float* __restrict__ outF, bf16* __restrict__ outB,
    int M, int N, int K)
{
  __shared__ bf16 As[128*32];    // [row][k]
  __shared__ bf16 Bst[128*32];   // transposed: [n][k]
  int tid  = threadIdx.x;
  int lane = tid & 31, wave = tid >> 5;
  int hi = lane >> 4, nl = lane & 15;
  int wm = (wave >> 1) * 32, wn = (wave & 1) * 64;
  int bm0 = blockIdx.x * 128, bn0 = blockIdx.y * 128;

  v8f acc[2][4];
  #pragma unroll
  for (int i = 0; i < 2; i++)
    #pragma unroll
    for (int j = 0; j < 4; j++) acc[i][j] = zero8();

  for (int k0 = 0; k0 < K; k0 += 32) {
#ifdef HAVE_TDM
    // wave 0: async TDM DMA of the 128x32 A tile into LDS
    if (tid < 32) {
      tdm_load_2d((unsigned)(size_t)As, A + (size_t)bm0*K + k0,
                  /*tile_w=*/32, /*tile_h=*/128,
                  /*tensor_w=*/(unsigned)K, /*tensor_h=*/(unsigned)M,
                  /*stride=*/(unsigned)K);
    }
#else
    #pragma unroll
    for (int c = 0; c < 2; c++) {
      int idx8 = tid + c*256;
      int r = idx8 >> 2, kc = (idx8 & 3) * 8;
      *(v8bf*)(As + r*32 + kc) = *(const v8bf*)(A + (size_t)(bm0 + r)*K + k0 + kc);
    }
#endif
    // all waves: cooperative B tile load (32x128), stored transposed
    #pragma unroll
    for (int c = 0; c < 2; c++) {
      int idx8 = tid + c*256;
      int kr = idx8 >> 4, nc = (idx8 & 15) * 8;
      v8bf bv = *(const v8bf*)(Bm + (size_t)(k0 + kr)*N + bn0 + nc);
      #pragma unroll
      for (int e = 0; e < 8; e++) Bst[(nc + e)*32 + kr] = bv[e];
    }
    if (k0 + 32 < K)   // gfx1250 global_prefetch_b8 for next B tile
      __builtin_prefetch(Bm + (size_t)(k0 + 32 + (tid >> 3))*N + bn0, 0, 1);
#ifdef HAVE_TDM
    if (tid < 32) __builtin_amdgcn_s_wait_tensorcnt(0);
#endif
    __syncthreads();

    v16bf aF[2], bF[4];
    #pragma unroll
    for (int i = 0; i < 2; i++) {
      const bf16* p = As + (wm + i*16 + nl)*32;
      aF[i] = frag_split(p + hi*16, p + hi*16 + 8);
    }
    #pragma unroll
    for (int j = 0; j < 4; j++)   // B frag: lane n holds k = hi*16 + j, contiguous in Bst
      bF[j] = *(const v16bf*)(Bst + (wn + j*16 + nl)*32 + hi*16);

    #pragma unroll
    for (int i = 0; i < 2; i++)
      #pragma unroll
      for (int j = 0; j < 4; j++)
        acc[i][j] = wmma_bf16(aF[i], bF[j], acc[i][j]);
    __syncthreads();
  }

  // epilogue: C layout m = r + hi*8, n = nl
  #pragma unroll
  for (int i = 0; i < 2; i++) {
    #pragma unroll
    for (int j = 0; j < 4; j++) {
      int gcol = bn0 + wn + j*16 + nl;
      float bz = bias ? bias[gcol] : 0.0f;
      #pragma unroll
      for (int r = 0; r < 8; r++) {
        int grow = bm0 + wm + i*16 + r + hi*8;
        size_t o = (size_t)grow * N + gcol;
        float vv = acc[i][j][r] + bz;
        if (RES)    vv += res[o];
        if (RELU)   vv = fmaxf(vv, 0.0f);
        if (F32OUT) outF[o] = vv;
        if (BF16OUT)outB[o] = (bf16)vv;
      }
    }
  }
}

// ---------------- flash attention (causal, online softmax) ----------------
// qkv: [BT, 3*D] bf16 (q|k|v per head). block = 4 waves, wave = 16 queries of one (b,h).

__global__ void __launch_bounds__(128) attn_kernel(const bf16* __restrict__ qkv,
                                                   bf16* __restrict__ o)
{
  __shared__ bf16 pS[4][16*32];   // per-wave P staging (C-layout -> A-layout via LDS)
  int tid  = threadIdx.x;
  int lane = tid & 31, wave = tid >> 5;
  int hi = lane >> 4, nl = lane & 15;
  int bh = blockIdx.x;
  int b = bh / HH, h = bh % HH;
  int rowbase = b * TT;
  int q0 = blockIdx.y * 64 + wave * 16;
  const int QO = h*HSZ, KO = DD + h*HSZ, VO = 2*DD + h*HSZ;
  bf16* pSw = pS[wave];

  // Q fragments: 16x64 = two K=32 A-frags
  v16bf aQ[2];
  #pragma unroll
  for (int c = 0; c < 2; c++) {
    const bf16* qp = qkv + (size_t)(rowbase + q0 + nl)*D3 + QO + c*32;
    aQ[c] = frag_split(qp + hi*16, qp + hi*16 + 8);
  }

  v8f aO[4];
  #pragma unroll
  for (int nt = 0; nt < 4; nt++) aO[nt] = zero8();
  float rowm[8], rowl[8];
  #pragma unroll
  for (int r = 0; r < 8; r++) { rowm[r] = -1e30f; rowl[r] = 0.0f; }

  for (int s0 = 0; s0 <= q0 + 15; s0 += 32) {
    // scores for two 16-key subtiles: S = Q*K^T (K-dim = HS = 64)
    v8f sc0 = zero8(), sc1 = zero8();
    #pragma unroll
    for (int c = 0; c < 2; c++) {
      v16bf bK0 = *(const v16bf*)(qkv + (size_t)(rowbase + s0      + nl)*D3 + KO + c*32 + hi*16);
      v16bf bK1 = *(const v16bf*)(qkv + (size_t)(rowbase + s0 + 16 + nl)*D3 + KO + c*32 + hi*16);
      sc0 = wmma_bf16(aQ[c], bK0, sc0);
      sc1 = wmma_bf16(aQ[c], bK1, sc1);
    }
    float alpha[8];
    #pragma unroll
    for (int r = 0; r < 8; r++) {
      int qrow = q0 + r + hi*8;
      float a0 = sc0[r] * 0.125f; if (s0      + nl > qrow) a0 = -1e30f;
      float a1 = sc1[r] * 0.125f; if (s0 + 16 + nl > qrow) a1 = -1e30f;
      // row max across the 16-lane group holding this row
      float vm = fmaxf(a0, a1);
      vm = fmaxf(vm, __shfl_xor(vm, 1, 32));
      vm = fmaxf(vm, __shfl_xor(vm, 2, 32));
      vm = fmaxf(vm, __shfl_xor(vm, 4, 32));
      vm = fmaxf(vm, __shfl_xor(vm, 8, 32));
      float newm = fmaxf(rowm[r], vm);
      alpha[r] = __expf(rowm[r] - newm);
      rowm[r] = newm;
      float p0 = __expf(a0 - newm);
      float p1 = __expf(a1 - newm);
      float ps = p0 + p1;
      ps += __shfl_xor(ps, 1, 32);
      ps += __shfl_xor(ps, 2, 32);
      ps += __shfl_xor(ps, 4, 32);
      ps += __shfl_xor(ps, 8, 32);
      rowl[r] = rowl[r] * alpha[r] + ps;
      pSw[(r + hi*8)*32 + nl]      = (bf16)p0;
      pSw[(r + hi*8)*32 + 16 + nl] = (bf16)p1;
    }
    #pragma unroll
    for (int nt = 0; nt < 4; nt++)
      #pragma unroll
      for (int r = 0; r < 8; r++) aO[nt][r] *= alpha[r];

    // O += P(16x32) * V(32x64)
    const bf16* pr = pSw + nl*32;
    v16bf aP = frag_split(pr + hi*16, pr + hi*16 + 8);
    #pragma unroll
    for (int nt = 0; nt < 4; nt++) {
      v16bf bV;
      #pragma unroll
      for (int j = 0; j < 16; j++)
        bV[j] = qkv[(size_t)(rowbase + s0 + hi*16 + j)*D3 + VO + nt*16 + nl];
      aO[nt] = wmma_bf16(aP, bV, aO[nt]);
    }
  }

  float inv[8];
  #pragma unroll
  for (int r = 0; r < 8; r++) inv[r] = 1.0f / rowl[r];
  #pragma unroll
  for (int nt = 0; nt < 4; nt++)
    #pragma unroll
    for (int r = 0; r < 8; r++)
      o[(size_t)(rowbase + q0 + r + hi*8)*DD + h*HSZ + nt*16 + nl] = (bf16)(aO[nt][r] * inv[r]);
}

// ---------------- host orchestration ----------------

extern "C" void kernel_launch(void* const* d_in, const int* in_sizes, int n_in,
                              void* d_out, int out_size, void* d_ws, size_t ws_size,
                              hipStream_t stream) {
  const int*   idx  = (const int*)  d_in[0];
  const float* tok  = (const float*)d_in[1];
  const float* pos  = (const float*)d_in[2];
  const float* Wq   = (const float*)d_in[3];
  const float* Wk   = (const float*)d_in[4];
  const float* Wv   = (const float*)d_in[5];
  const float* Wp   = (const float*)d_in[6];
  const float* bp   = (const float*)d_in[7];
  const float* ln1g = (const float*)d_in[8];
  const float* ln1b = (const float*)d_in[9];
  const float* ln2g = (const float*)d_in[10];
  const float* ln2b = (const float*)d_in[11];
  const float* W1   = (const float*)d_in[12];
  const float* b1   = (const float*)d_in[13];
  const float* W2   = (const float*)d_in[14];
  const float* b2   = (const float*)d_in[15];
  const float* lnfg = (const float*)d_in[16];
  const float* lnfb = (const float*)d_in[17];
  const float* lmW  = (const float*)d_in[18];
  const float* lmb  = (const float*)d_in[19];
  float* out = (float*)d_out;

  char* ws = (char*)d_ws;
  size_t off = 0;
  auto take = [&](size_t bytes) {
    char* p = ws + off;
    off = (off + bytes + 255) & ~(size_t)255;
    return p;
  };
  float* x   = (float*)take((size_t)BT*DD*4);
  bf16* hb   = (bf16*) take((size_t)BT*DD*2);
  bf16* qkvb = (bf16*) take((size_t)BT*D3*2);
  bf16* ob   = (bf16*) take((size_t)BT*DD*2);
  bf16* mb   = (bf16*) take((size_t)BT*D4*2);
  bf16* wqkv = (bf16*) take((size_t)LL*DD*D3*2);
  bf16* wp   = (bf16*) take((size_t)LL*DD*DD*2);
  bf16* w1   = (bf16*) take((size_t)LL*DD*D4*2);
  bf16* w2   = (bf16*) take((size_t)LL*D4*DD*2);
  bf16* lmw  = (bf16*) take((size_t)DD*VV*2);

  // bf16 weight arena (re-built every call; deterministic)
  int nq = LL*HH*DD*HSZ;
  cvt_qkv<<<nq/256, 256, 0, stream>>>(Wq, wqkv, 0);
  cvt_qkv<<<nq/256, 256, 0, stream>>>(Wk, wqkv, DD);
  cvt_qkv<<<nq/256, 256, 0, stream>>>(Wv, wqkv, 2*DD);
  cvt_flat<<<(LL*DD*DD)/256, 256, 0, stream>>>(Wp, wp, LL*DD*DD);
  cvt_flat<<<(LL*DD*D4)/256, 256, 0, stream>>>(W1, w1, LL*DD*D4);
  cvt_flat<<<(LL*D4*DD)/256, 256, 0, stream>>>(W2, w2, LL*D4*DD);
  cvt_flat<<<(DD*VV)/256, 256, 0, stream>>>(lmW, lmw, DD*VV);

  embed_kernel<<<BT, 256, 0, stream>>>(idx, tok, pos, x);

  for (int l = 0; l < LL; l++) {
    ln_kernel<<<BT, 256, 0, stream>>>(x, ln1g + l*DD, ln1b + l*DD, hb);
    gemm_kernel<false,false,false,true><<<dim3(BT/128, D3/128), 256, 0, stream>>>(
        hb, wqkv + (size_t)l*DD*D3, nullptr, nullptr, nullptr, qkvb, BT, D3, DD);
    attn_kernel<<<dim3(BB*HH, TT/64), 128, 0, stream>>>(qkvb, ob);
    gemm_kernel<false,true,true,false><<<dim3(BT/128, DD/128), 256, 0, stream>>>(
        ob, wp + (size_t)l*DD*DD, bp + l*DD, x, x, nullptr, BT, DD, DD);
    ln_kernel<<<BT, 256, 0, stream>>>(x, ln2g + l*DD, ln2b + l*DD, hb);
    gemm_kernel<true,false,false,true><<<dim3(BT/128, D4/128), 256, 0, stream>>>(
        hb, w1 + (size_t)l*DD*D4, b1 + l*D4, nullptr, nullptr, mb, BT, D4, DD);
    gemm_kernel<false,true,true,false><<<dim3(BT/128, DD/128), 256, 0, stream>>>(
        mb, w2 + (size_t)l*D4*DD, b2 + l*DD, x, x, nullptr, BT, DD, D4);
  }
  ln_kernel<<<BT, 256, 0, stream>>>(x, lnfg, lnfb, hb);
  gemm_kernel<false,false,true,false><<<dim3(BT/128, VV/128), 256, 0, stream>>>(
      hb, lmw, lmb, nullptr, out, nullptr, BT, VV, DD);

  (void)in_sizes; (void)n_in; (void)out_size; (void)ws_size;
}
